// VQVAE_seq_WT_46007689675074
// MI455X (gfx1250) — compile-verified, matrix operand order checked by reading, not run on hardware
//
#include <hip/hip_runtime.h>
#include <hip/hip_bf16.h>

// Problem constants
#define NUM_E 512        // codebook entries K
#define CDIM  128        // embedding dim C
#define SVOX  32768      // 32*32*32 spatial voxels per batch
#define NTOT  65536      // B * SVOX rows
#define NELEM 8388608.0f // total input elements (2*128*32768) for commit mean

typedef __attribute__((ext_vector_type(16))) __bf16 v16bf;
typedef __attribute__((ext_vector_type(8)))  float  v8f;

union BfFrag {
  v16bf v;
  __bf16 e[16];
  unsigned int u[8];
  uint4 q[2];
};

// d_out layout (floats), concatenated in reference return order:
// quantized [2,128,32,32,32], loss [1], indices [2,32,32,32] (as float),
// new_cluster_size [512], new_ema_w [512,128], new_embedding_w [512,128]
#define OFF_Q     0
#define OFF_LOSS  2097152
#define OFF_IDX   2097153
#define OFF_CS    2162689
#define OFF_EMAW  2163201
#define OFF_EMB   2228737

// ws layout: e2[512] floats, then pre-swizzled B fragments (bf16-pair dwords):
//   Bhi[32768] uints, Blo[32768] uints.  Fragment slot for (n, g, kc, v):
//   slot = ((n*2 + g)*4 + kc)*8 + v, holding packed {bf16(E[n][c]), bf16(E[n][c+1])}
//   with c = 32*kc + 16*(v>>2) + 8*g + 2*(v&3)   (ISA 16-bit B 32x16 map).
#define WS_BHI_OFF 512   // in floats/uints from ws base

__device__ __forceinline__ int frag_chan(int kc, int g, int v) {
  return 32 * kc + ((v >> 2) << 4) + (g << 3) + ((v & 3) << 1);
}

// ---------------------------------------------------------------------------
// Kernel 1: init accumulators in d_out, precompute ||e_k||^2 and the
// pre-swizzled bf16 hi/lo codebook fragments into ws.
// ---------------------------------------------------------------------------
__global__ __launch_bounds__(256) void vq_init(const float* __restrict__ E,
                                               const float* __restrict__ ema_cs,
                                               const float* __restrict__ ema_w,
                                               float* __restrict__ out,
                                               float* __restrict__ ws_e2,
                                               unsigned int* __restrict__ ws_bhi,
                                               unsigned int* __restrict__ ws_blo) {
  const int i = blockIdx.x * blockDim.x + threadIdx.x;

  if (i < NUM_E * CDIM) out[OFF_EMAW + i] = 0.99f * ema_w[i];

  if (i < NUM_E) {
    out[OFF_CS + i] = 0.99f * ema_cs[i];
    const float* row = E + i * CDIM;
    float s = 0.f;
    for (int c = 0; c < CDIM; ++c) s += row[c] * row[c];
    ws_e2[i] = s;
  }

  // Swizzle codebook into WMMA B-fragment layout (one dword slot per thread).
  if (i < NUM_E * 64) {
    const int v  = i & 7;
    const int kc = (i >> 3) & 3;
    const int g  = (i >> 5) & 1;
    const int n  = i >> 6;
    const int c  = frag_chan(kc, g, v);
    const float e0 = E[n * CDIM + c];
    const float e1 = E[n * CDIM + c + 1];
    const __bf16 h0 = (__bf16)e0, h1 = (__bf16)e1;
    const __bf16 l0 = (__bf16)(e0 - (float)h0), l1 = (__bf16)(e1 - (float)h1);
    union { __bf16 h[2]; unsigned int w; } ph, pl;
    ph.h[0] = h0; ph.h[1] = h1;
    pl.h[0] = l0; pl.h[1] = l1;
    ws_bhi[i] = ph.w;
    ws_blo[i] = pl.w;
  }

  if (i == 0) out[OFF_LOSS] = 0.f;
}

// ---------------------------------------------------------------------------
// Kernel 2: fused distances (split-bf16 WMMA) + online softmax/argmin +
//           quantized write + commit loss + EMA scatter.
// Block = 256 threads = 8 waves; each wave owns 16 rows of flat_input.
// ---------------------------------------------------------------------------
__global__ __launch_bounds__(256) void vq_main(const float* __restrict__ in,
                                               const float* __restrict__ E,
                                               float* __restrict__ out,
                                               const float* __restrict__ ws_e2,
                                               const unsigned int* __restrict__ ws_bhi,
                                               const unsigned int* __restrict__ ws_blo) {
  __shared__ float lds_e2[NUM_E];
  __shared__ float lds_dot[8][16][16];
  __shared__ int   lds_idx[8][16];

  const int t = threadIdx.x;
  lds_e2[t]       = ws_e2[t];
  lds_e2[t + 256] = ws_e2[t + 256];
  __syncthreads();

  const int wv   = t >> 5;
  const int lane = t & 31;
  const int m    = lane & 15;   // row within 16-row group / column within tile
  const int g    = lane >> 4;   // half-wave group (K-striping per ISA layout)

  const int row0 = blockIdx.x * 128 + wv * 16;
  const int b    = row0 >> 15;          // SVOX == 2^15
  const int s0   = row0 & (SVOX - 1);
  // flat_input[n][c] = in[b*C*S + c*S + (s0+m)]  (column-major in n)
  const float* X = in + (size_t)b * CDIM * SVOX + s0;

  // --- Preload A panel (16 rows x 128 channels) as bf16 hi + lo residual ---
  BfFrag a_hi[4], a_lo[4];
  float x2p = 0.f;
  for (int kc = 0; kc < 4; ++kc) {
    for (int v = 0; v < 8; ++v) {
      const int c  = frag_chan(kc, g, v);
      const float x0 = X[(size_t)c * SVOX + m];
      const float x1 = X[(size_t)(c + 1) * SVOX + m];
      const __bf16 h0 = (__bf16)x0, h1 = (__bf16)x1;
      a_hi[kc].e[2 * v]     = h0;
      a_hi[kc].e[2 * v + 1] = h1;
      a_lo[kc].e[2 * v]     = (__bf16)(x0 - (float)h0);
      a_lo[kc].e[2 * v + 1] = (__bf16)(x1 - (float)h1);
      x2p += x0 * x0 + x1 * x1;
    }
  }
  // each lane holds half of its row's channels; combine with the buddy lane
  const float x2 = x2p + __shfl_xor(x2p, 16);

  // --- Online softmax / argmin state (rows handled by lanes 0..15) ---
  float dmin = 3.0e37f, Zr = 0.f, Tr = 0.f;
  int   amin = 0;

  // Per-(column,group) fragment streams, 32 dwords per tile step.
  const unsigned int* __restrict__ bhiP = ws_bhi + ((size_t)m * 2 + g) * 32;
  const unsigned int* __restrict__ bloP = ws_blo + ((size_t)m * 2 + g) * 32;

  for (int kt = 0; kt < 32; ++kt) {
    const int n0 = kt << 4;
    // fragments for column n0+m: advance by 16 columns * 64 dwords each tile
    const unsigned int* bhi = bhiP + (size_t)n0 * 64;
    const unsigned int* blo = bloP + (size_t)n0 * 64;
    if (kt < 31) {  // prefetch next tile's fragments (global_prefetch_b8)
      __builtin_prefetch(bhi + 16 * 64, 0, 3);
      __builtin_prefetch(blo + 16 * 64, 0, 3);
    }

    v8f acc = {0.f, 0.f, 0.f, 0.f, 0.f, 0.f, 0.f, 0.f};
    for (int kc = 0; kc < 4; ++kc) {
      BfFrag bh, bl;
      bh.q[0] = ((const uint4*)(bhi + kc * 8))[0];
      bh.q[1] = ((const uint4*)(bhi + kc * 8))[1];
      bl.q[0] = ((const uint4*)(blo + kc * 8))[0];
      bl.q[1] = ((const uint4*)(blo + kc * 8))[1];
      // split-precision product: x.e ~= hh + hl + lh  (fp32-class accuracy)
      acc = __builtin_amdgcn_wmma_f32_16x16x32_bf16(false, a_hi[kc].v, false, bh.v,
                                                    (short)0, acc, false, false);
      acc = __builtin_amdgcn_wmma_f32_16x16x32_bf16(false, a_hi[kc].v, false, bl.v,
                                                    (short)0, acc, false, false);
      acc = __builtin_amdgcn_wmma_f32_16x16x32_bf16(false, a_lo[kc].v, false, bh.v,
                                                    (short)0, acc, false, false);
    }

    // C/D map: VGPR v holds row (v + 8*g), lane column = lane&15
    for (int v = 0; v < 8; ++v)
      lds_dot[wv][v + (g << 3)][m] = acc[v];
    __builtin_amdgcn_wave_barrier();   // in-order DS; fence compiler reordering

    if (lane < 16) {
      for (int j = 0; j < 16; ++j) {
        const float d = x2 + lds_e2[n0 + j] - 2.f * lds_dot[wv][lane][j];
        if (d < dmin) {                 // strict '<' => first-occurrence argmin
          const float delta = d - dmin; // rescale running sums to new max
          const float es = __expf(delta);
          Tr = es * (Tr + delta * Zr);
          Zr = es * Zr;
          dmin = d; amin = n0 + j;
        }
        const float u = dmin - d;       // y_j - max(y) <= 0
        const float w = __expf(u);
        Zr += w; Tr += u * w;
      }
    }
    __builtin_amdgcn_wave_barrier();
  }

  // --- per-row results: index, entropy, histogram ---
  if (lane < 16) {
    const int row = row0 + lane;
    out[OFF_IDX + row] = (float)amin;          // indices emitted as float
    lds_idx[wv][lane]  = amin;
    const float ent = __logf(Zr) - Tr / Zr;    // -sum p*log p for this row
    atomicAdd(&out[OFF_LOSS], (0.01f / (float)NTOT) * ent);
    atomicAdd(&out[OFF_CS + amin], 0.01f);     // 0.01 * encodings_sum
  }
  __builtin_amdgcn_wave_barrier();

  // --- quantized gather-write + commit loss + dw scatter (all 32 lanes) ---
  const int idx = lds_idx[wv][m];
  const float* Q = E + (size_t)idx * CDIM;
  const size_t base = (size_t)b * CDIM * SVOX + s0 + m;
  float commit = 0.f;
  for (int c = g; c < CDIM; c += 2) {          // lanes span 16 consecutive voxels
    const size_t o = base + (size_t)c * SVOX;
    const float q = Q[c];
    const float x = in[o];
    out[OFF_Q + o] = q;                        // straight-through fwd == q_raw
    const float df = q - x;
    commit += df * df;
    atomicAdd(&out[OFF_EMAW + idx * CDIM + c], 0.01f * x);  // 0.01 * dw
  }
  atomicAdd(&out[OFF_LOSS], (0.25f / NELEM) * commit);
}

// ---------------------------------------------------------------------------
// Kernel 3: n_total reduction + smoothing + new_embedding_w
// ---------------------------------------------------------------------------
__global__ __launch_bounds__(NUM_E) void vq_final(float* __restrict__ out) {
  __shared__ float red[NUM_E];
  const int k = threadIdx.x;
  const float ncs = out[OFF_CS + k];
  red[k] = ncs;
  __syncthreads();
  for (int s = NUM_E >> 1; s > 0; s >>= 1) {
    if (k < s) red[k] += red[k + s];
    __syncthreads();
  }
  const float n_total  = red[0];
  const float smoothed = (ncs + 1e-5f) / (n_total + NUM_E * 1e-5f) * n_total;
  const float inv = 1.f / smoothed;
  for (int c = 0; c < CDIM; ++c)
    out[OFF_EMB + k * CDIM + c] = out[OFF_EMAW + k * CDIM + c] * inv;
}

// ---------------------------------------------------------------------------
extern "C" void kernel_launch(void* const* d_in, const int* in_sizes, int n_in,
                              void* d_out, int out_size, void* d_ws, size_t ws_size,
                              hipStream_t stream) {
  (void)in_sizes; (void)n_in; (void)out_size; (void)ws_size;
  const float* in_vol = (const float*)d_in[0];  // [2,128,32,32,32] f32
  const float* E      = (const float*)d_in[1];  // [512,128] f32
  const float* emacs  = (const float*)d_in[2];  // [512] f32
  const float* emaw   = (const float*)d_in[3];  // [512,128] f32
  float* out = (float*)d_out;

  float* ws_e2 = (float*)d_ws;                              // 512 floats
  unsigned int* ws_bhi = (unsigned int*)ws_e2 + WS_BHI_OFF; // 32768 dwords
  unsigned int* ws_blo = ws_bhi + NUM_E * 64;               // 32768 dwords

  vq_init <<<256, 256, 0, stream>>>(E, emacs, emaw, out, ws_e2, ws_bhi, ws_blo);
  vq_main <<<512, 256, 0, stream>>>(in_vol, E, out, ws_e2, ws_bhi, ws_blo);
  vq_final<<<1, NUM_E, 0, stream>>>(out);
}
